// MultiHeadAttentionLayer2_85847806312756
// MI455X (gfx1250) — matrix operand order repaired and checked by reading, not compiled, not run on hardware
//
#include <hip/hip_runtime.h>
#include <hip/hip_bf16.h>

// ---------------------------------------------------------------------------
// MHA layer for MI455X (gfx1250, wave32, WMMA 16x16x32 f16 -> f32 accum).
//
// All reused fp32 GEMM operands are pre-converted to f16 ONCE, so every GEMM
// hot loop is pure global_load_b128 -> v_wmma (no per-iteration cvt).  The
// only fp32 hot-loop operand is the attention matrix (must live fp32 in
// d_out anyway).
//
// Register-blocked GEMM: each wave computes a 64x32 output macro-tile as
// 4x2 WMMA 16x16 subtiles (8 accumulators, A frags reused x2, B frags x4).
//
// All GEMMs expressed as D[m,n] = sum_k A[m,k] * Bt[n,k]  (K contiguous in
// both operands).  f16 intermediates / copies in d_ws (~95.4 MB):
//   Qh   [B, LQ, HID]   KcT/VcT [B, HID, LC]   Kn [B, LK, HID]
//   VnT  [B, HID, LK]   Xatt    [B, LQ, HID]
//   queryH/keyH/valueH  [B, L, HID]   WqH..WoH [HID, HID]   rWH [LK, LC]
// d_out = [x : B*LQ*HID f32][attention : B*NH*LQ*LK f32]
// ---------------------------------------------------------------------------

#define DEV __device__ __forceinline__

typedef _Float16 half4_t __attribute__((ext_vector_type(4)));
typedef _Float16 half8_t __attribute__((ext_vector_type(8)));
typedef _Float16 v16h    __attribute__((ext_vector_type(16)));
typedef float    v8f     __attribute__((ext_vector_type(8)));
typedef float    f4v     __attribute__((ext_vector_type(4)));

enum { SRC_F32 = 0, SRC_F16 = 1 };
enum { OUT_F32 = 0, OUT_F16 = 1, OUT_F16T = 2 };
enum { BIAS_NONE = 0, BIAS_N = 1, BIAS_M = 2 };

// --- A fragment: 16x32 f16, lane<16 holds row m, K {kc+0..7, kc+16..23};
//     lane>=16 holds row m, K {kc+8..15, kc+24..31}   (ISA 7.12.2)
DEV v16h load_a_f16(const _Float16* __restrict__ A, long lda, int m, int kc, bool lo) {
  const _Float16* p = A + (long)m * lda + kc;
  half8_t x0 = *(const half8_t*)(p + (lo ? 0 : 8));
  half8_t x1 = *(const half8_t*)(p + (lo ? 16 : 24));
  v16h a;
#pragma unroll
  for (int i = 0; i < 8; ++i) { a[i] = x0[i]; a[i + 8] = x1[i]; }
  return a;
}

DEV v16h load_a_f32(const float* __restrict__ A, long lda, int m, int kc, bool lo) {
  const float* p = A + (long)m * lda + kc;
  const int g0 = lo ? 0 : 8, g1 = lo ? 16 : 24;
  f4v x0 = *(const f4v*)(p + g0);
  f4v x1 = *(const f4v*)(p + g0 + 4);
  f4v x2 = *(const f4v*)(p + g1);
  f4v x3 = *(const f4v*)(p + g1 + 4);
  v16h a;
#pragma unroll
  for (int i = 0; i < 4; ++i) {
    a[i]      = (_Float16)x0[i];
    a[i + 4]  = (_Float16)x1[i];
    a[i + 8]  = (_Float16)x2[i];
    a[i + 12] = (_Float16)x3[i];
  }
  return a;
}

// --- B fragment: 32x16 f16, lane holds column n = lane&15;
//     lane<16: K {kc+0..7, kc+16..23}; lane>=16: K {kc+8..15, kc+24..31}; Bt is [N,K] row-major.
DEV v16h load_b_f16(const _Float16* __restrict__ Bt, long ldb, int n, int kc, bool lo) {
  const _Float16* p = Bt + (long)n * ldb + kc + (lo ? 0 : 8);
  half8_t x0 = *(const half8_t*)(p);
  half8_t x1 = *(const half8_t*)(p + 16);
  v16h b;
#pragma unroll
  for (int i = 0; i < 8; ++i) { b[i] = x0[i]; b[i + 8] = x1[i]; }
  return b;
}

// ---------------------------------------------------------------------------
// Register-blocked wave-tile WMMA GEMM: each wave owns a 64x32 macro-tile
// (4 M-subtiles x 2 N-subtiles).  grid.x covers macro-tiles (4 waves/block),
// grid.y / grid.z select batch / head via the sAy/sAz etc. strides.
// Requires M % 64 == 0 and N % 32 == 0 (true for all call sites).
// ---------------------------------------------------------------------------
template <int ASRC, int OUTM, int BIASM>
__launch_bounds__(128)
__global__ void wmma_gemm(const void* __restrict__ Ap, long lda, long sAy, long sAz,
                          const _Float16* __restrict__ Bp, long ldb, long sBy, long sBz,
                          void* __restrict__ Dp, long ldd, long sDy, long sDz,
                          const float* __restrict__ bias,
                          int M, int N, int K, float scale) {
  const int lane = threadIdx.x & 31;
  const bool lo  = lane < 16;
  const int tilesN = N >> 5;  // 32-wide macro-tiles
  const int tilesM = M >> 6;  // 64-tall macro-tiles
  const int tile = blockIdx.x * (blockDim.x >> 5) + (threadIdx.x >> 5);
  if (tile >= tilesM * tilesN) return;  // wave-uniform: EXEC stays all-ones
  const int tm = (tile / tilesN) << 6;
  const int tn = (tile % tilesN) << 5;

  const long offA = (long)blockIdx.y * sAy + (long)blockIdx.z * sAz;
  const long offB = (long)blockIdx.y * sBy + (long)blockIdx.z * sBz;
  const long offD = (long)blockIdx.y * sDy + (long)blockIdx.z * sDz;

  const _Float16* A16 = (const _Float16*)Ap + offA;
  const float*    A32 = (const float*)Ap + offA;
  const _Float16* B16 = Bp + offB;

  const int l15 = lane & 15;

  v8f acc[4][2] = {};
  for (int kc = 0; kc < K; kc += 32) {
    if (kc + 32 < K) {  // prefetch next K-chunk of A rows (global_prefetch_b8)
      const int mp = tm + l15;
      if (ASRC == SRC_F16) __builtin_prefetch(A16 + (long)mp * lda + kc + 32, 0, 0);
      else                 __builtin_prefetch(A32 + (long)mp * lda + kc + 32, 0, 0);
    }
    v16h a[4], b[2];
#pragma unroll
    for (int i = 0; i < 4; ++i) {
      const int mi = tm + i * 16 + l15;
      a[i] = (ASRC == SRC_F16) ? load_a_f16(A16, lda, mi, kc, lo)
                               : load_a_f32(A32, lda, mi, kc, lo);
    }
#pragma unroll
    for (int j = 0; j < 2; ++j) {
      const int nj = tn + j * 16 + l15;
      b[j] = load_b_f16(B16, ldb, nj, kc, lo);
    }
#pragma unroll
    for (int i = 0; i < 4; ++i)
#pragma unroll
      for (int j = 0; j < 2; ++j)
        acc[i][j] = __builtin_amdgcn_wmma_f32_16x16x32_f16(
            false, a[i], false, b[j], (short)0, acc[i][j], false, false);
  }

  // C/D layout per subtile: VGPR r, lane<16 -> M=r, lane>=16 -> M=8+r; N=lane&15
#pragma unroll
  for (int i = 0; i < 4; ++i) {
    const int mbase = tm + i * 16 + (lo ? 0 : 8);
#pragma unroll
    for (int j = 0; j < 2; ++j) {
      const int nj = tn + j * 16 + l15;
      if (OUTM == OUT_F32) {
        float* D = (float*)Dp + offD;
#pragma unroll
        for (int r = 0; r < 8; ++r) {
          float v = acc[i][j][r] * scale;
          if (BIASM == BIAS_N) v += bias[nj];
          if (BIASM == BIAS_M) v += bias[mbase + r];
          D[(long)(mbase + r) * ldd + nj] = v;
        }
      } else if (OUTM == OUT_F16) {
        _Float16* D = (_Float16*)Dp + offD;
#pragma unroll
        for (int r = 0; r < 8; ++r) {
          float v = acc[i][j][r] * scale;
          if (BIASM == BIAS_N) v += bias[nj];
          if (BIASM == BIAS_M) v += bias[mbase + r];
          D[(long)(mbase + r) * ldd + nj] = (_Float16)v;
        }
      } else {  // OUT_F16T: D[n, m] -- contiguous 8 halves per lane
        _Float16* D = (_Float16*)Dp + offD;
        half8_t v8x;
#pragma unroll
        for (int r = 0; r < 8; ++r) {
          float v = acc[i][j][r] * scale;
          if (BIASM == BIAS_N) v += bias[nj];
          if (BIASM == BIAS_M) v += bias[mbase + r];
          v8x[r] = (_Float16)v;
        }
        *(half8_t*)(D + (long)nj * ldd + mbase) = v8x;
      }
    }
  }
}

// ---------------------------------------------------------------------------
// One-time fp32 -> f16 conversion (vectorized x4; n must be a multiple of 4).
// ---------------------------------------------------------------------------
__global__ void f32_to_f16(const float* __restrict__ src,
                           _Float16* __restrict__ dst, long n) {
  long i = ((long)blockIdx.x * blockDim.x + threadIdx.x) * 4;
  if (i >= n) return;
  f4v x = *(const f4v*)(src + i);
  half4_t h;
#pragma unroll
  for (int r = 0; r < 4; ++r) h[r] = (_Float16)x[r];
  *(half4_t*)(dst + i) = h;
}

// ---------------------------------------------------------------------------
// sampled [B, LS, HID] f32  ->  KcT[b][h][s] and VcT[b][h][s] f16 (s < LS)
// ---------------------------------------------------------------------------
__global__ void pack_sampled(const float* __restrict__ s,
                             _Float16* __restrict__ KcT,
                             _Float16* __restrict__ VcT,
                             int Bn, int LSn, int H, int LCn) {
  long idx = (long)blockIdx.x * blockDim.x + threadIdx.x;
  long total = (long)Bn * LSn * H;
  if (idx >= total) return;
  int h = (int)(idx % H);
  long t = idx / H;
  int ss = (int)(t % LSn);
  int b  = (int)(t / LSn);
  _Float16 v = (_Float16)s[idx];
  long o = ((long)b * H + h) * LCn + ss;
  KcT[o] = v;
  VcT[o] = v;
}

// ---------------------------------------------------------------------------
// Row softmax in place: one 256-thread block per row of length L.
// ---------------------------------------------------------------------------
__launch_bounds__(256)
__global__ void softmax_rows(float* __restrict__ x, int L) {
  float* row = x + (long)blockIdx.x * L;
  __shared__ float red[256];
  const int t = threadIdx.x;

  float mx = -3.4e38f;
  for (int i = t; i < L; i += 256) mx = fmaxf(mx, row[i]);
  red[t] = mx;
  __syncthreads();
  for (int s2 = 128; s2 > 0; s2 >>= 1) {
    if (t < s2) red[t] = fmaxf(red[t], red[t + s2]);
    __syncthreads();
  }
  mx = red[0];
  __syncthreads();

  float sum = 0.f;
  for (int i = t; i < L; i += 256) {
    float e = __expf(row[i] - mx);
    row[i] = e;
    sum += e;
  }
  red[t] = sum;
  __syncthreads();
  for (int s2 = 128; s2 > 0; s2 >>= 1) {
    if (t < s2) red[t] += red[t + s2];
    __syncthreads();
  }
  float inv = 1.f / red[0];
  for (int i = t; i < L; i += 256) row[i] *= inv;
}

static inline dim3 gemm_grid(int M, int N, int y, int z) {
  int tiles = (M / 64) * (N / 32);       // 64x32 macro-tile per wave
  int blocks = (tiles + 3) / 4;          // 4 waves per 128-thread block
  return dim3((unsigned)blocks, (unsigned)y, (unsigned)z);
}

static inline void cvt(const float* s, _Float16* d, long n, hipStream_t st) {
  f32_to_f16<<<(unsigned)((n / 4 + 255) / 256), 256, 0, st>>>(s, d, n);
}

extern "C" void kernel_launch(void* const* d_in, const int* in_sizes, int n_in,
                              void* d_out, int out_size, void* d_ws, size_t ws_size,
                              hipStream_t stream) {
  (void)in_sizes; (void)n_in; (void)out_size; (void)ws_size;

  const int Bc = 2, LQ = 2048, LK = 2048, LS = 256, HID = 1024, NH = 16, HD = 64;
  const int LC = LS + LK;  // 2304

  const float* query   = (const float*)d_in[0];
  const float* sampled = (const float*)d_in[1];
  const float* key     = (const float*)d_in[2];
  const float* value   = (const float*)d_in[3];
  const float* Wq = (const float*)d_in[4];
  const float* bq = (const float*)d_in[5];
  const float* Wk = (const float*)d_in[6];
  const float* bk = (const float*)d_in[7];
  const float* Wv = (const float*)d_in[8];
  const float* bv = (const float*)d_in[9];
  const float* Wo = (const float*)d_in[10];
  const float* bo = (const float*)d_in[11];
  const float* rW = (const float*)d_in[12];
  const float* rb = (const float*)d_in[13];

  // workspace carve-out (f16), ~95.4 MB total
  _Float16* Qh     = (_Float16*)d_ws;
  _Float16* KcT    = Qh     + (size_t)Bc * LQ * HID;
  _Float16* VcT    = KcT    + (size_t)Bc * HID * LC;
  _Float16* Kn     = VcT    + (size_t)Bc * HID * LC;
  _Float16* VnT    = Kn     + (size_t)Bc * LK * HID;
  _Float16* Xatt   = VnT    + (size_t)Bc * HID * LK;
  _Float16* queryH = Xatt   + (size_t)Bc * LQ * HID;
  _Float16* keyH   = queryH + (size_t)Bc * LQ * HID;
  _Float16* valueH = keyH   + (size_t)Bc * LK * HID;
  _Float16* WqH    = valueH + (size_t)Bc * LK * HID;
  _Float16* WkH    = WqH    + (size_t)HID * HID;
  _Float16* WvH    = WkH    + (size_t)HID * HID;
  _Float16* WoH    = WvH    + (size_t)HID * HID;
  _Float16* rWH    = WoH    + (size_t)HID * HID;

  float* outX = (float*)d_out;                       // [B, LQ, HID]
  float* attn = outX + (size_t)Bc * LQ * HID;        // [B, NH, LQ, LK]

  // 0) one-time fp32 -> f16 conversions of all reused GEMM operands
  cvt(query, queryH, (long)Bc * LQ * HID, stream);
  cvt(key,   keyH,   (long)Bc * LK * HID, stream);
  cvt(value, valueH, (long)Bc * LK * HID, stream);
  cvt(Wq, WqH, (long)HID * HID, stream);
  cvt(Wk, WkH, (long)HID * HID, stream);
  cvt(Wv, WvH, (long)HID * HID, stream);
  cvt(Wo, WoH, (long)HID * HID, stream);
  cvt(rW, rWH, (long)LK * LC, stream);

  // 1) sampled -> transposed f16 prefix of Kc and Vc
  {
    long total = (long)Bc * LS * HID;
    pack_sampled<<<(unsigned)((total + 255) / 256), 256, 0, stream>>>(
        sampled, KcT, VcT, Bc, LS, HID, LC);
  }

  // 2) Q = query @ Wq^T + bq  -> Qh f16 [B, LQ, HID]
  wmma_gemm<SRC_F16, OUT_F16, BIAS_N>
      <<<gemm_grid(LQ, HID, Bc, 1), 128, 0, stream>>>(
          queryH, HID, (long)LQ * HID, 0,
          WqH, HID, 0, 0,
          Qh, HID, (long)LQ * HID, 0,
          bq, LQ, HID, HID, 1.0f);

  // 3) K = key @ Wk^T + bk  -> transposed into KcT[:, :, LS:]
  wmma_gemm<SRC_F16, OUT_F16T, BIAS_N>
      <<<gemm_grid(LK, HID, Bc, 1), 128, 0, stream>>>(
          keyH, HID, (long)LK * HID, 0,
          WkH, HID, 0, 0,
          KcT + LS, LC, (long)HID * LC, 0,
          bk, LK, HID, HID, 1.0f);

  // 4) V = value @ Wv^T + bv -> transposed into VcT[:, :, LS:]
  wmma_gemm<SRC_F16, OUT_F16T, BIAS_N>
      <<<gemm_grid(LK, HID, Bc, 1), 128, 0, stream>>>(
          valueH, HID, (long)LK * HID, 0,
          WvH, HID, 0, 0,
          VcT + LS, LC, (long)HID * LC, 0,
          bv, LK, HID, HID, 1.0f);

  // 5) K' = rW @ Kc + rb  -> Kn f16 [B, LK, HID]
  wmma_gemm<SRC_F16, OUT_F16, BIAS_M>
      <<<gemm_grid(LK, HID, Bc, 1), 128, 0, stream>>>(
          rWH, LC, 0, 0,
          KcT, LC, (long)HID * LC, 0,
          Kn, HID, (long)LK * HID, 0,
          rb, LK, HID, LC, 1.0f);

  // 6) V' = rW @ Vc + rb  -> transposed VnT f16 [B, HID, LK]
  wmma_gemm<SRC_F16, OUT_F16T, BIAS_M>
      <<<gemm_grid(LK, HID, Bc, 1), 128, 0, stream>>>(
          rWH, LC, 0, 0,
          VcT, LC, (long)HID * LC, 0,
          VnT, LK, (long)HID * LK, 0,
          rb, LK, HID, LC, 1.0f);

  // 7) energy = Qh @ Kh^T / sqrt(HD)  -> attn region of d_out (f32)
  wmma_gemm<SRC_F16, OUT_F32, BIAS_NONE>
      <<<gemm_grid(LQ, LK, Bc, NH), 128, 0, stream>>>(
          Qh, HID, (long)LQ * HID, HD,
          Kn, HID, (long)LK * HID, HD,
          attn, LK, (long)NH * LQ * LK, (long)LQ * LK,
          nullptr, LQ, LK, HD, 0.125f);

  // 8) softmax over last axis, in place
  softmax_rows<<<(unsigned)(Bc * NH * LQ), 256, 0, stream>>>(attn, LK);

  // 9) x = attn @ Vh  -> Xatt f16 [B, LQ, HID]   (A stays fp32: lives in d_out)
  wmma_gemm<SRC_F32, OUT_F16, BIAS_NONE>
      <<<gemm_grid(LQ, HD, Bc, NH), 128, 0, stream>>>(
          attn, LK, (long)NH * LQ * LK, (long)LQ * LK,
          VnT, LK, (long)HID * LK, (long)HD * LK,
          Xatt, HID, (long)LQ * HID, HD,
          nullptr, LQ, HD, LK, 1.0f);

  // 10) out = Xatt @ Wo^T + bo -> f32 d_out x-region
  wmma_gemm<SRC_F16, OUT_F32, BIAS_N>
      <<<gemm_grid(LQ, HID, Bc, 1), 128, 0, stream>>>(
          Xatt, HID, (long)LQ * HID, 0,
          WoH, HID, 0, 0,
          outX, HID, (long)LQ * HID, 0,
          bo, LQ, HID, HID, 1.0f);
}
